// Brain_49374944035335
// MI455X (gfx1250) — matrix operand order repaired and checked by reading, not compiled, not run on hardware
//
#include <hip/hip_runtime.h>

// ---------------------------------------------------------------------------
// CDNA5 (gfx1250) implementation of the Brain reference:
//  - convs as im2col + f16 WMMA GEMM (v_wmma_f32_16x16x32_f16, f32 accum)
//  - batched per-node MLPs on the same WMMA GEMM (grid.z = 16 nodes)
//  - A-tile staging via global_load_async_to_lds_b128 (ASYNCcnt)
//  - B-tile staging via b128 global loads + packed-b32 transposed LDS stores
// Activations are CNHW f16. Requires ~285MB workspace.
// ---------------------------------------------------------------------------

typedef __attribute__((ext_vector_type(16))) _Float16 v16h;
typedef __attribute__((ext_vector_type(8)))  float    v8f;

#define GEMM_BLK_M 128
#define GEMM_BLK_N 256
#define GEMM_BLK_K 32
#define LDS_PAD    40   // padded row length (f16) => 80B rows, 16B aligned

struct alignas(16) U4 { unsigned x, y, z, w; };
union AFrag { unsigned u[8]; v16h h; };           // 32B: A 16x32 f16 fragment
union BFrag { U4 q[2];      v16h h; };            // 32B: B 32x16 f16 fragment
union Q16   { U4 q;         _Float16 h[8]; };     // 16B reinterpret helper
union P2    { _Float16 h[2]; unsigned u; };       // f16 pair <-> dword

// ---------------------------------------------------------------------------
// Generic batched WMMA GEMM:  C[b] = op( A[b](MxK,f16) * B[b](KxN,f16) + bias )
// ep: 0 = none, 1 = relu, 2 = tanh.  C stored f16 row-major [M][N].
// ---------------------------------------------------------------------------
__launch_bounds__(256)
__global__ void k_gemm_wmma(const _Float16* __restrict__ Aall,
                            const _Float16* __restrict__ Ball,
                            const float*    __restrict__ biasAll,
                            _Float16*       __restrict__ Call,
                            int M, int N, int K,
                            long sA, long sB, long sBias, long sC, int ep)
{
    __shared__ __align__(16) _Float16 At[GEMM_BLK_M][LDS_PAD];
    __shared__ __align__(16) _Float16 Bt[GEMM_BLK_N][LDS_PAD];

    const int tid   = threadIdx.x;
    const int lane  = tid & 31;
    const int wave  = tid >> 5;        // 0..7
    const int waveM = wave >> 2;       // 0..1  (64-row strip)
    const int waveN = wave & 3;        // 0..3  (64-col strip)
    const int half  = lane >> 4;       // lane group
    const int l16   = lane & 15;

    const int batch = blockIdx.z;
    const _Float16* A    = Aall    + (long)batch * sA;
    const _Float16* Bm   = Ball    + (long)batch * sB;
    const float*    bias = biasAll + (long)batch * sBias;
    _Float16*       C    = Call    + (long)batch * sC;

    const int m0 = blockIdx.y * GEMM_BLK_M;
    const int n0 = blockIdx.x * GEMM_BLK_N;
    const bool nfull = (n0 + GEMM_BLK_N <= N);

    v8f acc[4][4];
    const v8f vz = {};
    #pragma unroll
    for (int i = 0; i < 4; ++i)
        #pragma unroll
        for (int j = 0; j < 4; ++j) acc[i][j] = vz;

    for (int k0 = 0; k0 < K; k0 += GEMM_BLK_K) {
        const bool kfull = (k0 + GEMM_BLK_K <= K);

        // ---- stage A tile [128][32]: thread t -> row t/2, half t&1 (16 f16 = 32B)
        {
            const int r  = tid >> 1;
            const int hh = tid & 1;
            const int gm = m0 + r;
            if (kfull) {
                if (gm < M) {
                    // async DMA: 32B global -> LDS (offset applies to both sides)
                    const unsigned lds_off = (unsigned)(size_t)&At[r][hh * 16];
                    const unsigned goff    = (unsigned)(((long)gm * K + k0 + hh * 16) * 2);
                    asm volatile(
                        "global_load_async_to_lds_b128 %0, %1, %2\n\t"
                        "global_load_async_to_lds_b128 %0, %1, %2 offset:16"
                        :: "v"(lds_off), "v"(goff),
                           "s"((unsigned long long)(size_t)A)
                        : "memory");
                } else {
                    const U4 z = {0u, 0u, 0u, 0u};
                    *(U4*)&At[r][hh * 16]     = z;
                    *(U4*)&At[r][hh * 16 + 8] = z;
                }
            } else {
                #pragma unroll
                for (int kk = 0; kk < 16; ++kk) {
                    const int gk = k0 + hh * 16 + kk;
                    _Float16 v = (_Float16)0.f;
                    if (gm < M && gk < K) v = A[(long)gm * K + gk];
                    At[r][hh * 16 + kk] = v;
                }
            }
        }

        // ---- stage B transposed [256][32]
        if (kfull && nfull) {
            // wide path: thread t -> K pair (2*kkp) x 16-col segment
            const int kkp = tid >> 4;        // 0..15
            const int seg = tid & 15;        // 0..15
            const long gk = (long)k0 + 2 * kkp;
            const _Float16* src0 = Bm + gk * N + n0 + seg * 16;
            const _Float16* src1 = src0 + N;
            Q16 qa0, qa1, qb0, qb1;
            qa0.q = *(const U4*)(src0);
            qa1.q = *(const U4*)(src0 + 8);
            qb0.q = *(const U4*)(src1);
            qb1.q = *(const U4*)(src1 + 8);
            #pragma unroll
            for (int j = 0; j < 16; ++j) {
                P2 p;
                p.h[0] = (j < 8) ? qa0.h[j & 7] : qa1.h[j & 7];   // K = gk
                p.h[1] = (j < 8) ? qb0.h[j & 7] : qb1.h[j & 7];   // K = gk+1
                *(unsigned*)&Bt[seg * 16 + j][2 * kkp] = p.u;
            }
        } else {
            // edge path: thread t -> column n0+t, scalar guarded
            const int gn = n0 + tid;
            #pragma unroll 4
            for (int kk = 0; kk < GEMM_BLK_K; ++kk) {
                const int gk = k0 + kk;
                _Float16 v = (_Float16)0.f;
                if (gn < N && gk < K) v = Bm[(long)gk * N + gn];
                Bt[tid][kk] = v;
            }
        }

        // drain async A-tile DMAs, then block barrier
        asm volatile("s_wait_asynccnt 0" ::: "memory");
        __syncthreads();

        // ---- load fragments, contiguous k = 16*half + e in both A and B
        AFrag af[4];
        BFrag bf[4];
        #pragma unroll
        for (int mi = 0; mi < 4; ++mi) {
            const int row = waveM * 64 + mi * 16 + l16;   // A row = lane%16
            #pragma unroll
            for (int j = 0; j < 8; ++j) {
                // K pair base: 16*lanehalf + 2*j
                const int kb = 16 * half + (j << 1);
                af[mi].u[j] = *(const unsigned*)&At[row][kb];
            }
        }
        #pragma unroll
        for (int ni = 0; ni < 4; ++ni) {
            const int col = waveN * 64 + ni * 16 + l16;   // B col = lane%16
            const _Float16* p = &Bt[col][half * 16];      // K = half*16 + e
            bf[ni].q[0] = *(const U4*)(p);
            bf[ni].q[1] = *(const U4*)(p + 8);
        }

        // ---- 16 WMMAs per wave per 32-K step
        #pragma unroll
        for (int mi = 0; mi < 4; ++mi)
            #pragma unroll
            for (int ni = 0; ni < 4; ++ni)
                acc[mi][ni] = __builtin_amdgcn_wmma_f32_16x16x32_f16(
                    false, af[mi].h, false, bf[ni].h,
                    (short)0, acc[mi][ni], false, false);

        __syncthreads();
    }

    // ---- epilogue: C/D layout: VGPR r -> M = r + 8*(lane/16), N = lane%16
    #pragma unroll
    for (int mi = 0; mi < 4; ++mi) {
        #pragma unroll
        for (int ni = 0; ni < 4; ++ni) {
            #pragma unroll
            for (int r = 0; r < 8; ++r) {
                const int gm = m0 + waveM * 64 + mi * 16 + r + 8 * half;
                const int gn = n0 + waveN * 64 + ni * 16 + l16;
                if (gm < M && gn < N) {
                    float v = acc[mi][ni][r] + bias[gm];
                    if (ep == 1)      v = v > 0.f ? v : 0.f;
                    else if (ep == 2) v = tanhf(v);
                    C[(long)gm * N + gn] = (_Float16)v;
                }
            }
        }
    }
}

// ---------------------------------------------------------------------------
// Elementwise / data-movement kernels
// ---------------------------------------------------------------------------
__global__ void k_f32_to_f16(const float* __restrict__ s, _Float16* __restrict__ d, long n)
{
    long i = (long)blockIdx.x * blockDim.x + threadIdx.x;
    if (i < n) d[i] = (_Float16)s[i];
}

// x NCHW f32 -> CNHW f16
__global__ void k_x_cnhw(const float* __restrict__ x, _Float16* __restrict__ d,
                         int B, int C, int HW)
{
    long total = (long)B * C * HW;
    long i = (long)blockIdx.x * blockDim.x + threadIdx.x;
    if (i >= total) return;
    int hw = (int)(i % HW);
    long t = i / HW;
    int c = (int)(t % C);
    int b = (int)(t / C);
    d[((long)c * B + b) * HW + hw] = (_Float16)x[i];
}

// act CNHW f16 -> col [C*9][B*H*W] f16 (3x3 SAME)
__global__ void k_im2col(const _Float16* __restrict__ act, _Float16* __restrict__ col,
                         int B, int H, int W)
{
    const long N = (long)B * H * W;
    long n = (long)blockIdx.x * blockDim.x + threadIdx.x;
    const int k = blockIdx.y;          // c*9 + r*3 + s
    if (n >= N) return;
    const int c = k / 9, rs = k % 9, r = rs / 3, s = rs % 3;
    const int HWl = H * W;
    const int b = (int)(n / HWl);
    const int rem = (int)(n % HWl);
    const int y = rem / W, xx = rem % W;
    const int yi = y + r - 1, xi = xx + s - 1;
    _Float16 v = (_Float16)0.f;
    if (yi >= 0 && yi < H && xi >= 0 && xi < W)
        v = act[((long)c * B + b) * HWl + (long)yi * W + xi];
    col[(long)k * N + n] = v;
}

// per-channel batch stats (train-mode BN): block = channel
__global__ void k_bn_stats(const _Float16* __restrict__ act, float* __restrict__ mean,
                           float* __restrict__ rstd, long N)
{
    const int c = blockIdx.x;
    const _Float16* p = act + (long)c * N;
    float s = 0.f, s2 = 0.f;
    for (long i = threadIdx.x; i < N; i += blockDim.x) {
        float v = (float)p[i];
        s += v; s2 += v * v;
    }
    __shared__ float sh[256], sh2[256];
    sh[threadIdx.x] = s; sh2[threadIdx.x] = s2;
    __syncthreads();
    for (int o = 128; o > 0; o >>= 1) {
        if (threadIdx.x < o) { sh[threadIdx.x] += sh[threadIdx.x + o];
                               sh2[threadIdx.x] += sh2[threadIdx.x + o]; }
        __syncthreads();
    }
    if (threadIdx.x == 0) {
        float m = sh[0] / (float)N;
        float var = sh2[0] / (float)N - m * m;
        mean[c] = m;
        rstd[c] = rsqrtf(var + 1e-5f);
    }
}

__global__ void k_bn_relu(_Float16* __restrict__ act, const float* __restrict__ mean,
                          const float* __restrict__ rstd, const float* __restrict__ g,
                          const float* __restrict__ b, long N)
{
    const int c = blockIdx.y;
    long i = (long)blockIdx.x * blockDim.x + threadIdx.x;
    if (i >= N) return;
    const long idx = (long)c * N + i;
    float v = (float)act[idx];
    v = (v - mean[c]) * rstd[c] * g[c] + b[c];
    act[idx] = (_Float16)(v > 0.f ? v : 0.f);
}

// maxpool k3 s2 p1 on CNHW; grid.y = channel
__global__ void k_maxpool(const _Float16* __restrict__ in, _Float16* __restrict__ out,
                          int B, int H, int W)
{
    const int c = blockIdx.y;
    const int Ho = H >> 1, Wo = W >> 1;
    const long Nout = (long)B * Ho * Wo;
    long i = (long)blockIdx.x * blockDim.x + threadIdx.x;
    if (i >= Nout) return;
    const int b = (int)(i / (Ho * Wo));
    const int rem = (int)(i % (Ho * Wo));
    const int y = rem / Wo, xx = rem % Wo;
    float m = -1e30f;
    #pragma unroll
    for (int dy = -1; dy <= 1; ++dy)
        #pragma unroll
        for (int dx = -1; dx <= 1; ++dx) {
            const int yi = 2 * y + dy, xi = 2 * xx + dx;
            if (yi >= 0 && yi < H && xi >= 0 && xi < W) {
                float v = (float)in[((long)c * B + b) * H * W + (long)yi * W + xi];
                m = v > m ? v : m;
            }
        }
    out[((long)c * B + b) * Ho * Wo + (long)y * Wo + xx] = (_Float16)m;
}

// X[n][b][592]: first 80 zero, [80:592] = act7[c][b][n] (act7 = [512][B][16])
__global__ void k_build_X(const _Float16* __restrict__ act7, _Float16* __restrict__ X, int B)
{
    const long total = (long)16 * B * 592;
    long i = (long)blockIdx.x * blockDim.x + threadIdx.x;
    if (i >= total) return;
    const int f = (int)(i % 592);
    const long nb = i / 592;
    const int b = (int)(nb % B);
    const int n = (int)(nb / B);
    _Float16 v = (_Float16)0.f;
    if (f >= 80) v = act7[((long)(f - 80) * B + b) * 16 + n];
    X[i] = v;
}

// exact reference neighbor ordering on a 4x4 grid, 8-connectivity
__device__ int neigh_list(int i, int out[8])
{
    const int w = 4, size = 16;
    int cnt = 0;
    if (i - w >= 0)                         out[cnt++] = i - w;
    if (i % w != 0)                         out[cnt++] = i - 1;
    if ((i + 1) % w != 0)                   out[cnt++] = i + 1;
    if (i + w < size)                       out[cnt++] = i + w;
    if (i - w - 1 >= 0 && i % w != 0)       out[cnt++] = i - w - 1;
    if (i - w + 1 >= 0 && (i + 1) % w != 0) out[cnt++] = i - w + 1;
    if (i + w - 1 < size && i % w != 0)     out[cnt++] = i + w - 1;
    if (i + w + 1 < size && (i + 1) % w != 0) out[cnt++] = i + w + 1;
    return cnt;
}

// overwrite X[n][b][0:80] with gathered neighbor preds (zero padded)
__global__ void k_gather(const float* __restrict__ preds1, _Float16* __restrict__ X, int B)
{
    const long total = (long)16 * B * 80;
    long i = (long)blockIdx.x * blockDim.x + threadIdx.x;
    if (i >= total) return;
    const int f = (int)(i % 80);
    const long nb = i / 80;
    const int b = (int)(nb % B);
    const int n = (int)(nb / B);
    const int j = f / 10, cls = f % 10;
    int nl[8];
    const int cnt = neigh_list(n, nl);
    float v = 0.f;
    if (j < cnt) v = preds1[((long)nl[j] * B + b) * 10 + cls];
    X[((long)n * B + b) * 592 + f] = (_Float16)v;
}

// softmax over 10 classes; one thread per (node, batch)
__global__ void k_softmax(const _Float16* __restrict__ logits, float* __restrict__ probs, int NB)
{
    int i = blockIdx.x * blockDim.x + threadIdx.x;
    if (i >= NB) return;
    const _Float16* p = logits + (long)i * 10;
    float v[10], mx = -1e30f;
    #pragma unroll
    for (int c = 0; c < 10; ++c) { v[c] = (float)p[c]; mx = v[c] > mx ? v[c] : mx; }
    float s = 0.f;
    #pragma unroll
    for (int c = 0; c < 10; ++c) { v[c] = __expf(v[c] - mx); s += v[c]; }
    const float inv = 1.f / s;
    #pragma unroll
    for (int c = 0; c < 10; ++c) probs[(long)i * 10 + c] = v[c] * inv;
}

// squared mean over nodes
__global__ void k_final(const float* __restrict__ preds2, float* __restrict__ out, int B)
{
    int i = blockIdx.x * blockDim.x + threadIdx.x;
    if (i >= B * 10) return;
    const int b = i / 10, c = i % 10;
    float s = 0.f;
    #pragma unroll
    for (int n = 0; n < 16; ++n) s += preds2[((long)n * B + b) * 10 + c];
    s *= (1.f / 16.f);
    out[i] = s * s;
}

// ---------------------------------------------------------------------------
// Host-side orchestration
// ---------------------------------------------------------------------------
static inline char* wsoff(char* base, size_t& off, size_t bytes)
{
    char* p = base + off;
    off += (bytes + 255) & ~(size_t)255;
    return p;
}

static void launch_gemm(const _Float16* A, const _Float16* Bm, const float* bias,
                        _Float16* C, int M, int N, int K,
                        long sA, long sB, long sBias, long sC, int ep, int batch,
                        hipStream_t stream)
{
    dim3 g((N + GEMM_BLK_N - 1) / GEMM_BLK_N, (M + GEMM_BLK_M - 1) / GEMM_BLK_M, batch);
    k_gemm_wmma<<<g, 256, 0, stream>>>(A, Bm, bias, C, M, N, K, sA, sB, sBias, sC, ep);
}

static void cvt(const float* s, _Float16* d, long n, hipStream_t stream)
{
    k_f32_to_f16<<<(unsigned)((n + 255) / 256), 256, 0, stream>>>(s, d, n);
}

extern "C" void kernel_launch(void* const* d_in, const int* in_sizes, int n_in,
                              void* d_out, int out_size, void* d_ws, size_t ws_size,
                              hipStream_t stream)
{
    (void)in_sizes; (void)n_in; (void)out_size; (void)ws_size;
    const int B = 128;

    const float* x     = (const float*)d_in[0];
    const float* cw1   = (const float*)d_in[1];
    const float* cb1   = (const float*)d_in[2];
    const float* bn1_g = (const float*)d_in[3];
    const float* bn1_b = (const float*)d_in[4];
    const float* cw2   = (const float*)d_in[5];
    const float* cb2   = (const float*)d_in[6];
    const float* bn2_g = (const float*)d_in[7];
    const float* bn2_b = (const float*)d_in[8];
    const float* cw3   = (const float*)d_in[9];
    const float* cb3   = (const float*)d_in[10];
    const float* cw4   = (const float*)d_in[11];
    const float* cb4   = (const float*)d_in[12];
    const float* cw5   = (const float*)d_in[13];
    const float* cb5   = (const float*)d_in[14];
    const float* cw6   = (const float*)d_in[15];
    const float* cb6   = (const float*)d_in[16];
    const float* cw7   = (const float*)d_in[17];
    const float* cb7   = (const float*)d_in[18];
    const float* W1    = (const float*)d_in[19];
    const float* b1    = (const float*)d_in[20];
    const float* W2    = (const float*)d_in[21];
    const float* b2    = (const float*)d_in[22];
    float* out = (float*)d_out;

    // ---- workspace layout (~285 MB) ----
    char* base = (char*)d_ws;
    size_t off = 0;
    _Float16* wsW    = (_Float16*)wsoff(base, off, (size_t)12637632 * 2);   // all f16 weights
    _Float16* actA   = (_Float16*)wsoff(base, off, (size_t)9000000 * 2);    // act ping
    _Float16* actB   = (_Float16*)wsoff(base, off, (size_t)9000000 * 2);    // act pong
    _Float16* wsOut  = (_Float16*)wsoff(base, off, (size_t)34000000 * 2);   // gemm out / act
    _Float16* wsCol  = (_Float16*)wsoff(base, off, (size_t)75497472 * 2);   // im2col
    _Float16* Xbuf   = (_Float16*)wsoff(base, off, (size_t)16 * B * 592 * 2);
    _Float16* Hbuf   = (_Float16*)wsoff(base, off, (size_t)16 * B * 600 * 2);
    _Float16* Lbuf   = (_Float16*)wsoff(base, off, (size_t)16 * B * 10 * 2);
    float*    preds1 = (float*)wsoff(base, off, (size_t)16 * B * 10 * 4);
    float*    meanB  = (float*)wsoff(base, off, (size_t)512 * 4);
    float*    rstdB  = meanB + 256;

    // f16 weight sub-regions
    _Float16* w1h = wsW;               // 64*27
    _Float16* w2h = w1h + 1728;        // 128*576
    _Float16* w3h = w2h + 73728;       // 256*1152
    _Float16* w4h = w3h + 294912;      // 256*2304
    _Float16* w5h = w4h + 589824;      // 512*2304
    _Float16* w6h = w5h + 1179648;     // 512*4608
    _Float16* w7h = w6h + 2359296;     // 512*4608
    _Float16* W1h = w7h + 2359296;     // 16*592*600
    _Float16* W2h = W1h + 5683200;     // 16*600*10

    cvt(cw1, w1h, 1728, stream);
    cvt(cw2, w2h, 73728, stream);
    cvt(cw3, w3h, 294912, stream);
    cvt(cw4, w4h, 589824, stream);
    cvt(cw5, w5h, 1179648, stream);
    cvt(cw6, w6h, 2359296, stream);
    cvt(cw7, w7h, 2359296, stream);
    cvt(W1,  W1h, 5683200, stream);
    cvt(W2,  W2h, 96000, stream);

    // ---- input to CNHW f16 ----
    {
        long tot = (long)B * 3 * 4096;
        k_x_cnhw<<<(unsigned)((tot + 255) / 256), 256, 0, stream>>>(x, actA, B, 3, 4096);
    }

    // ---- conv1: 3->64, 64x64, BN + relu + pool ----
    {
        const int Cin = 3, Cout = 64, H = 64, W = 64;
        const long N = (long)B * H * W;                              // 524288
        k_im2col<<<dim3((unsigned)((N + 255) / 256), Cin * 9), 256, 0, stream>>>(actA, wsCol, B, H, W);
        launch_gemm(w1h, wsCol, cb1, wsOut, Cout, (int)N, Cin * 9, 0, 0, 0, 0, /*ep=*/0, 1, stream);
        k_bn_stats<<<Cout, 256, 0, stream>>>(wsOut, meanB, rstdB, N);
        k_bn_relu<<<dim3((unsigned)((N + 255) / 256), Cout), 256, 0, stream>>>(wsOut, meanB, rstdB, bn1_g, bn1_b, N);
        k_maxpool<<<dim3((unsigned)(((long)B * 32 * 32 + 255) / 256), Cout), 256, 0, stream>>>(wsOut, actA, B, H, W);
    }
    // ---- conv2: 64->128, 32x32, BN + relu + pool ----
    {
        const int Cin = 64, Cout = 128, H = 32, W = 32;
        const long N = (long)B * H * W;                              // 131072
        k_im2col<<<dim3((unsigned)((N + 255) / 256), Cin * 9), 256, 0, stream>>>(actA, wsCol, B, H, W);
        launch_gemm(w2h, wsCol, cb2, wsOut, Cout, (int)N, Cin * 9, 0, 0, 0, 0, 0, 1, stream);
        k_bn_stats<<<Cout, 256, 0, stream>>>(wsOut, meanB, rstdB, N);
        k_bn_relu<<<dim3((unsigned)((N + 255) / 256), Cout), 256, 0, stream>>>(wsOut, meanB, rstdB, bn2_g, bn2_b, N);
        k_maxpool<<<dim3((unsigned)(((long)B * 16 * 16 + 255) / 256), Cout), 256, 0, stream>>>(wsOut, actB, B, H, W);
    }
    // ---- conv3: 128->256, 16x16, relu ----
    {
        const int Cin = 128, Cout = 256, H = 16, W = 16;
        const long N = (long)B * H * W;                              // 32768
        k_im2col<<<dim3((unsigned)((N + 255) / 256), Cin * 9), 256, 0, stream>>>(actB, wsCol, B, H, W);
        launch_gemm(w3h, wsCol, cb3, wsOut, Cout, (int)N, Cin * 9, 0, 0, 0, 0, /*relu*/1, 1, stream);
    }
    // ---- conv4: 256->256, 16x16, relu + pool ----
    {
        const int Cin = 256, Cout = 256, H = 16, W = 16;
        const long N = (long)B * H * W;
        k_im2col<<<dim3((unsigned)((N + 255) / 256), Cin * 9), 256, 0, stream>>>(wsOut, wsCol, B, H, W);
        launch_gemm(w4h, wsCol, cb4, wsOut, Cout, (int)N, Cin * 9, 0, 0, 0, 0, 1, 1, stream);
        k_maxpool<<<dim3((unsigned)(((long)B * 8 * 8 + 255) / 256), Cout), 256, 0, stream>>>(wsOut, actA, B, H, W);
    }
    // ---- conv5: 256->512, 8x8, relu ----
    {
        const int Cin = 256, Cout = 512, H = 8, W = 8;
        const long N = (long)B * H * W;                              // 8192
        k_im2col<<<dim3((unsigned)((N + 255) / 256), Cin * 9), 256, 0, stream>>>(actA, wsCol, B, H, W);
        launch_gemm(w5h, wsCol, cb5, wsOut, Cout, (int)N, Cin * 9, 0, 0, 0, 0, 1, 1, stream);
    }
    // ---- conv6: 512->512, 8x8, relu + pool ----
    {
        const int Cin = 512, Cout = 512, H = 8, W = 8;
        const long N = (long)B * H * W;
        k_im2col<<<dim3((unsigned)((N + 255) / 256), Cin * 9), 256, 0, stream>>>(wsOut, wsCol, B, H, W);
        launch_gemm(w6h, wsCol, cb6, wsOut, Cout, (int)N, Cin * 9, 0, 0, 0, 0, 1, 1, stream);
        k_maxpool<<<dim3((unsigned)(((long)B * 4 * 4 + 255) / 256), Cout), 256, 0, stream>>>(wsOut, actB, B, H, W);
    }
    // ---- conv7: 512->512, 4x4, relu -> act7 = wsOut [512][B][16] ----
    {
        const int Cin = 512, Cout = 512, H = 4, W = 4;
        const long N = (long)B * H * W;                              // 2048
        k_im2col<<<dim3((unsigned)((N + 255) / 256), Cin * 9), 256, 0, stream>>>(actB, wsCol, B, H, W);
        launch_gemm(w7h, wsCol, cb7, wsOut, Cout, (int)N, Cin * 9, 0, 0, 0, 0, 1, 1, stream);
    }

    // ---- colons, pass 1 (neighbor slots zero) ----
    {
        long tot = (long)16 * B * 592;
        k_build_X<<<(unsigned)((tot + 255) / 256), 256, 0, stream>>>(wsOut, Xbuf, B);
    }
    launch_gemm(Xbuf, W1h, b1, Hbuf, B, 600, 592,
                (long)B * 592, (long)592 * 600, 600, (long)B * 600, /*tanh*/2, 16, stream);
    launch_gemm(Hbuf, W2h, b2, Lbuf, B, 10, 600,
                (long)B * 600, (long)600 * 10, 10, (long)B * 10, 0, 16, stream);
    k_softmax<<<(16 * B + 255) / 256, 256, 0, stream>>>(Lbuf, preds1, 16 * B);

    // ---- colons, pass 2 (gather neighbor predictions) ----
    {
        long tot = (long)16 * B * 80;
        k_gather<<<(unsigned)((tot + 255) / 256), 256, 0, stream>>>(preds1, Xbuf, B);
    }
    launch_gemm(Xbuf, W1h, b1, Hbuf, B, 600, 592,
                (long)B * 592, (long)592 * 600, 600, (long)B * 600, 2, 16, stream);
    launch_gemm(Hbuf, W2h, b2, Lbuf, B, 10, 600,
                (long)B * 600, (long)600 * 10, 10, (long)B * 10, 0, 16, stream);

    float* preds2 = out + 1280;                    // [16][B][10] in return order after [B][10]
    k_softmax<<<(16 * B + 255) / 256, 256, 0, stream>>>(Lbuf, preds2, 16 * B);
    k_final<<<(B * 10 + 255) / 256, 256, 0, stream>>>(preds2, out, B);
}
